// PointTransformerCls_20907900797441
// MI455X (gfx1250) — matrix operand order repaired and checked by reading, not compiled, not run on hardware
//
#include <hip/hip_runtime.h>
#include <hip/hip_bf16.h>
#include <math.h>

// ---------------------------------------------------------------------------
// WMMA types (CDNA5 gfx1250, wave32)
// ---------------------------------------------------------------------------
typedef __attribute__((ext_vector_type(16))) _Float16 v16h;
typedef __attribute__((ext_vector_type(8)))  _Float16 v8h;
typedef __attribute__((ext_vector_type(4)))  _Float16 h4;
typedef __attribute__((ext_vector_type(8)))  float    v8f;
typedef __attribute__((ext_vector_type(4)))  float    f4;

union AF { v16h v; v8h h[2]; _Float16 e[16]; };
union CF { v8f  v; float e[8]; };

#define BM 128
#define BN 128
#define BK 32
#define ASTR 40   // halves; 80B row stride (16B aligned)
#define BSTR 40   // halves; sB is column-major: sB[n*BSTR + k]

// ---------------------------------------------------------------------------
// Specialized WMMA GEMM.  C[m,n] = sum_k Ae(m,k)*Be(k,n) (+bias, +=C opt.)
// MODE 0 (NT):   Ae = A[m*lda+k]            Be = B[n*ldb+k]       (x @ W^T)
// MODE 1 (TN):   Ae = A[k*lda+m]            Be = B[k*ldb+n]       (A^T @ B)
// MODE 2 (CONV): Ae = A[pm*lda+k], pm=(m/m_div)*m_stride+m%m_div+m_off
//                Be = B[n*ldb+k*bsk]
// All K are multiples of 32 (no K guards). M/N guarded per thread/element.
// 256 threads = 8 waves; 128x128 tile; each wave 64x32 -> 8 wmma / K-step.
// ---------------------------------------------------------------------------
template<int MODE>
__global__ __launch_bounds__(256) void k_gemm(
    const float* __restrict__ A, const float* __restrict__ Bm,
    const float* __restrict__ bias, float* __restrict__ C,
    int M, int N, int K, int lda, int ldb, int bsk, int ldc,
    int m_div, int m_stride, int m_off,
    long long sAb, long long sBb, long long sCb, int accumulate)
{
  __shared__ __align__(16) _Float16 sA[BM * ASTR];
  __shared__ __align__(16) _Float16 sB[BN * BSTR];

  A  += (long long)blockIdx.z * sAb;
  Bm += (long long)blockIdx.z * sBb;
  C  += (long long)blockIdx.z * sCb;

  const int tid  = threadIdx.x;
  const int wave = tid >> 5, lane = tid & 31;
  const int wr = wave >> 2;          // 0..1 : 64-row group
  const int wc = wave & 3;           // 0..3 : 32-col group
  const int m0 = blockIdx.y * BM;
  const int n0 = blockIdx.x * BN;
  const int crow = lane >> 4, ccol = lane & 15;

  CF acc[4][2];
  for (int ti = 0; ti < 4; ++ti)
    for (int tj = 0; tj < 2; ++tj)
      for (int r = 0; r < 8; ++r) {
        float cv = 0.f;
        if (accumulate) {
          int gm = m0 + wr * 64 + ti * 16 + crow * 8 + r;
          int gn = n0 + wc * 32 + tj * 16 + ccol;
          if (gm < M && gn < N) cv = C[(long long)gm * ldc + gn];
        }
        acc[ti][tj].e[r] = cv;
      }

  for (int kk = 0; kk < K; kk += BK) {
    // ---------------- stage A (BM x BK) ----------------
    if (MODE == 0 || MODE == 2) {
      int r = tid >> 1, half = tid & 1;          // 2 threads/row, 16 k each
      int m = m0 + r;
      _Float16* dst = &sA[r * ASTR + half * 16];
      if (m < M) {
        long long pm = (MODE == 2)
            ? (long long)(m / m_div) * m_stride + (m % m_div) + m_off
            : (long long)m;
        const float* src = A + pm * lda + kk + half * 16;
        if (kk + BK < K) __builtin_prefetch(src + BK);
        for (int j = 0; j < 4; ++j) {
          f4 f = *(const f4*)(src + 4 * j);
          h4 h; h.x = (_Float16)f.x; h.y = (_Float16)f.y;
                h.z = (_Float16)f.z; h.w = (_Float16)f.w;
          *(h4*)(dst + 4 * j) = h;
        }
      } else {
        for (int j = 0; j < 16; ++j) dst[j] = (_Float16)0.f;
      }
    } else { // MODE 1: A[k*lda + m], contiguous in m; scatter into sA
      int k = tid >> 3, mq = (tid & 7) * 16;
      const float* src = A + (long long)(kk + k) * lda + m0 + mq;
      if (m0 + mq + 16 <= M) {
        for (int j = 0; j < 4; ++j) {
          f4 f = *(const f4*)(src + 4 * j);
          sA[(mq + 4 * j + 0) * ASTR + k] = (_Float16)f.x;
          sA[(mq + 4 * j + 1) * ASTR + k] = (_Float16)f.y;
          sA[(mq + 4 * j + 2) * ASTR + k] = (_Float16)f.z;
          sA[(mq + 4 * j + 3) * ASTR + k] = (_Float16)f.w;
        }
      } else {
        for (int j = 0; j < 16; ++j) {
          float v = (m0 + mq + j < M) ? src[j] : 0.f;
          sA[(mq + j) * ASTR + k] = (_Float16)v;
        }
      }
    }
    // ---------------- stage B (BK x BN) into column-major sB ----------------
    if (MODE == 0) {
      int nl = tid >> 1, half = tid & 1;
      int n = n0 + nl;
      _Float16* dst = &sB[nl * BSTR + half * 16];
      if (n < N) {
        const float* src = Bm + (long long)n * ldb + kk + half * 16;
        if (kk + BK < K) __builtin_prefetch(src + BK);
        for (int j = 0; j < 4; ++j) {
          f4 f = *(const f4*)(src + 4 * j);
          h4 h; h.x = (_Float16)f.x; h.y = (_Float16)f.y;
                h.z = (_Float16)f.z; h.w = (_Float16)f.w;
          *(h4*)(dst + 4 * j) = h;
        }
      } else {
        for (int j = 0; j < 16; ++j) dst[j] = (_Float16)0.f;
      }
    } else if (MODE == 2) {  // strided-k weight slice (conv, bsk=3)
      int nl = tid >> 1, half = tid & 1;
      int n = n0 + nl;
      _Float16* dst = &sB[nl * BSTR + half * 16];
      if (n < N) {
        const float* src = Bm + (long long)n * ldb + (long long)(kk + half * 16) * bsk;
        for (int j = 0; j < 16; ++j) dst[j] = (_Float16)src[(long long)j * bsk];
      } else {
        for (int j = 0; j < 16; ++j) dst[j] = (_Float16)0.f;
      }
    } else { // MODE 1: B[k*ldb + n], contiguous in n; scatter into sB
      int k = tid >> 3, nq = (tid & 7) * 16;
      const float* src = Bm + (long long)(kk + k) * ldb + n0 + nq;
      if (n0 + nq + 16 <= N) {
        for (int j = 0; j < 4; ++j) {
          f4 f = *(const f4*)(src + 4 * j);
          sB[(nq + 4 * j + 0) * BSTR + k] = (_Float16)f.x;
          sB[(nq + 4 * j + 1) * BSTR + k] = (_Float16)f.y;
          sB[(nq + 4 * j + 2) * BSTR + k] = (_Float16)f.z;
          sB[(nq + 4 * j + 3) * BSTR + k] = (_Float16)f.w;
        }
      } else {
        for (int j = 0; j < 16; ++j) {
          float v = (n0 + nq + j < N) ? src[j] : 0.f;
          sB[(nq + j) * BSTR + k] = (_Float16)v;
        }
      }
    }
    __syncthreads();

    // ---------------- fragments (aligned 16B LDS loads) ----------------
    AF af[4], bf[2];
    const int arow = lane & 15, akoff = (lane >> 4) * 8;
    for (int ti = 0; ti < 4; ++ti) {
      const _Float16* pa = &sA[(wr * 64 + ti * 16 + arow) * ASTR];
      af[ti].h[0] = *(const v8h*)(pa + akoff);
      af[ti].h[1] = *(const v8h*)(pa + akoff + 16);
    }
    const int bkoff = (lane >> 4) * 8;
    for (int tj = 0; tj < 2; ++tj) {
      const _Float16* pb = &sB[(wc * 32 + tj * 16 + ccol) * BSTR + bkoff];
      bf[tj].h[0] = *(const v8h*)(pb);
      bf[tj].h[1] = *(const v8h*)(pb + 16);
    }
    for (int ti = 0; ti < 4; ++ti)
      for (int tj = 0; tj < 2; ++tj)
        acc[ti][tj].v = __builtin_amdgcn_wmma_f32_16x16x32_f16(
            false, af[ti].v, false, bf[tj].v, (short)0, acc[ti][tj].v, false, false);
    __syncthreads();
  }

  for (int ti = 0; ti < 4; ++ti)
    for (int tj = 0; tj < 2; ++tj)
      for (int r = 0; r < 8; ++r) {
        int gm = m0 + wr * 64 + ti * 16 + crow * 8 + r;
        int gn = n0 + wc * 32 + tj * 16 + ccol;
        if (gm < M && gn < N) {
          float v = acc[ti][tj].e[r];
          if (bias) v += bias[gn];
          C[(long long)gm * ldc + gn] = v;
        }
      }
}

// ---------------------------------------------------------------------------
// Support kernels
// ---------------------------------------------------------------------------
__global__ void k_linear3(const float* __restrict__ x, const float* __restrict__ w,
                          float* __restrict__ y, size_t rows) {
  size_t i = (size_t)blockIdx.x * 256 + threadIdx.x;
  if (i >= rows * 64) return;
  int o = (int)(i % 64); size_t r = i / 64;
  const float* px = x + r * 3; const float* pw = w + o * 3;
  y[i] = px[0]*pw[0] + px[1]*pw[1] + px[2]*pw[2];
}

__global__ void k_bn_stats(const float* __restrict__ X, float* __restrict__ mean,
                           float* __restrict__ var, int M, int C) {
  int c = blockIdx.x;
  __shared__ float s1[256], s2[256];
  float a = 0.f, b = 0.f;
  for (int r = threadIdx.x; r < M; r += 256) {
    float v = X[(size_t)r * C + c]; a += v; b += v * v;
  }
  s1[threadIdx.x] = a; s2[threadIdx.x] = b; __syncthreads();
  for (int off = 128; off > 0; off >>= 1) {
    if (threadIdx.x < off) { s1[threadIdx.x] += s1[threadIdx.x+off]; s2[threadIdx.x] += s2[threadIdx.x+off]; }
    __syncthreads();
  }
  if (threadIdx.x == 0) {
    float m = s1[0] / M; mean[c] = m; var[c] = s2[0] / M - m * m;
  }
}

__global__ void k_bn_act(const float* __restrict__ X, float* __restrict__ Y,
                         const float* __restrict__ mean, const float* __restrict__ var,
                         const float* __restrict__ g, const float* __restrict__ b,
                         size_t total, int C, int act) {
  size_t i = (size_t)blockIdx.x * 256 + threadIdx.x;
  if (i >= total) return;
  int c = (int)(i % C);
  float y = g[c] * (X[i] - mean[c]) * rsqrtf(var[c] + 1e-5f) + b[c];
  if (act == 1)      y = y / (1.f + expf(-y));        // SiLU
  else if (act == 2) y = y > 0.f ? y : 0.2f * y;      // leaky 0.2
  Y[i] = y;
}

__global__ void k_layernorm(float* __restrict__ X, const float* __restrict__ g,
                            const float* __restrict__ b, size_t M, int C) {
  size_t r = (size_t)blockIdx.x * 256 + threadIdx.x;
  if (r >= M) return;
  float* p = X + r * C;
  float s = 0.f, q = 0.f;
  for (int c = 0; c < C; ++c) { float v = p[c]; s += v; q += v * v; }
  float m = s / C, vv = q / C - m * m, inv = rsqrtf(vv + 1e-5f);
  for (int c = 0; c < C; ++c) p[c] = g[c] * (p[c] - m) * inv + b[c];
}

__global__ void k_fps(const float* __restrict__ xyz, int* __restrict__ out,
                      int Npts, int S) {
  int b = blockIdx.x;
  const float* p = xyz + (size_t)b * Npts * 3;
  __shared__ float dist[2048];
  __shared__ float rmax[256]; __shared__ int rarg[256];
  int tid = threadIdx.x;
  for (int i = tid; i < Npts; i += 256) dist[i] = 1e10f;
  __syncthreads();
  int far = 0;
  for (int s = 0; s < S; ++s) {
    if (tid == 0) out[b * S + s] = far;
    float cx = p[far*3], cy = p[far*3+1], cz = p[far*3+2];
    float lmax = -1.f; int larg = 0;
    for (int i = tid; i < Npts; i += 256) {
      float dx = p[i*3]-cx, dy = p[i*3+1]-cy, dz = p[i*3+2]-cz;
      float d = fminf(dist[i], dx*dx + dy*dy + dz*dz);
      dist[i] = d;
      if (d > lmax) { lmax = d; larg = i; }
    }
    rmax[tid] = lmax; rarg[tid] = larg;
    __syncthreads();
    for (int off = 128; off > 0; off >>= 1) {
      if (tid < off && rmax[tid+off] > rmax[tid]) { rmax[tid] = rmax[tid+off]; rarg[tid] = rarg[tid+off]; }
      __syncthreads();
    }
    far = rarg[0];
    __syncthreads();
  }
}

__global__ void k_knn(const float* __restrict__ d, int* __restrict__ knn,
                      int Npts, int KS) {
  int row = blockIdx.x;
  const float* dr = d + (size_t)row * Npts;
  __shared__ float v[2048];
  __shared__ float rmin[256]; __shared__ int rarg[256];
  int tid = threadIdx.x;
  for (int i = tid; i < Npts; i += 256) v[i] = dr[i];
  __syncthreads();
  for (int ks = 0; ks < KS; ++ks) {
    float lmin = 3.4e38f; int larg = 0;
    for (int i = tid; i < Npts; i += 256)
      if (v[i] < lmin) { lmin = v[i]; larg = i; }
    rmin[tid] = lmin; rarg[tid] = larg;
    __syncthreads();
    for (int off = 128; off > 0; off >>= 1) {
      if (tid < off && rmin[tid+off] < rmin[tid]) { rmin[tid] = rmin[tid+off]; rarg[tid] = rarg[tid+off]; }
      __syncthreads();
    }
    int sel = rarg[0];
    if (tid == 0) knn[(size_t)row * KS + ks] = sel;
    __syncthreads();
    v[sel] = 3.4e38f;
    __syncthreads();
  }
}

__global__ void k_sqdist(const float* __restrict__ a, const float* __restrict__ x,
                         float* __restrict__ d, int S, int Npts, size_t total) {
  size_t i = (size_t)blockIdx.x * 256 + threadIdx.x;
  if (i >= total) return;
  int n = (int)(i % Npts); size_t bs = i / Npts; int s = (int)(bs % S); int b = (int)(bs / S);
  const float* pa = a + ((size_t)b * S + s) * 3;
  const float* px = x + ((size_t)b * Npts + n) * 3;
  float dx = pa[0]-px[0], dy = pa[1]-px[1], dz = pa[2]-px[2];
  d[i] = dx*dx + dy*dy + dz*dz;
}

__global__ void k_gather3(const float* __restrict__ xyz, const int* __restrict__ idx,
                          float* __restrict__ out, int S, int Nin, size_t total) {
  size_t i = (size_t)blockIdx.x * 256 + threadIdx.x;
  if (i >= total) return;                 // total = B*S
  int b = (int)(i / S);
  int j = idx[i];
  const float* p = xyz + ((size_t)b * Nin + j) * 3;
  out[i*3] = p[0]; out[i*3+1] = p[1]; out[i*3+2] = p[2];
}

__global__ void k_gatherF(const float* __restrict__ feat, const int* __restrict__ idx,
                          float* __restrict__ out, int S, int Nin, int C, size_t total) {
  size_t i = (size_t)blockIdx.x * 256 + threadIdx.x;
  if (i >= total) return;                 // total = B*S*C
  int c = (int)(i % C); size_t bs = i / C; int b = (int)(bs / S);
  int j = idx[bs];
  out[i] = feat[((size_t)b * Nin + j) * C + c];
}

__global__ void k_group(const float* __restrict__ feat, const int* __restrict__ knn,
                        const float* __restrict__ center, float* __restrict__ nf,
                        int S, int KS, int C, int Nin, size_t total) {
  size_t i = (size_t)blockIdx.x * 256 + threadIdx.x;
  if (i >= total) return;                 // total = B*S*KS*C
  int c = (int)(i % C); size_t t = i / C; int k = (int)(t % KS);
  size_t bs = t / KS; int b = (int)(bs / S);
  int j = knn[bs * KS + k];
  float fv = feat[((size_t)b * Nin + j) * C + c];
  float cv = center[bs * C + c];
  float* o = nf + t * (size_t)(2 * C);
  o[c] = fv - cv; o[C + c] = cv;
}

__global__ void k_maxk(const float* __restrict__ X, float* __restrict__ out,
                       int KS, int C, size_t total) {
  size_t i = (size_t)blockIdx.x * 256 + threadIdx.x;
  if (i >= total) return;                 // total = B*S*C
  int c = (int)(i % C); size_t bs = i / C;
  const float* p = X + bs * (size_t)KS * C + c;
  float m = -3.4e38f;
  for (int k = 0; k < KS; ++k) m = fmaxf(m, p[(size_t)k * C]);
  out[i] = m;
}

__global__ void k_softmax256(float* __restrict__ E) {
  int row = blockIdx.x;
  float* p = E + (size_t)row * 256;
  __shared__ float sh[256];
  int t = threadIdx.x;
  float v = p[t];
  sh[t] = v; __syncthreads();
  for (int off = 128; off > 0; off >>= 1) {
    if (t < off) sh[t] = fmaxf(sh[t], sh[t+off]);
    __syncthreads();
  }
  float mx = sh[0]; __syncthreads();
  float e = expf(v - mx);
  sh[t] = e; __syncthreads();
  for (int off = 128; off > 0; off >>= 1) {
    if (t < off) sh[t] += sh[t+off];
    __syncthreads();
  }
  p[t] = e / sh[0];
}

__global__ void k_colsum(const float* __restrict__ E, float* __restrict__ cols,
                         int Np, size_t total) {
  size_t i = (size_t)blockIdx.x * 256 + threadIdx.x;
  if (i >= total) return;                 // total = B*Np ; (b,j)
  int j = (int)(i % Np); size_t b = i / Np;
  const float* p = E + b * (size_t)Np * Np + j;
  float s = 0.f;
  for (int r = 0; r < Np; ++r) s += p[(size_t)r * Np];
  cols[i] = s;
}

__global__ void k_coldiv(float* __restrict__ E, const float* __restrict__ cols,
                         int Np, size_t total) {
  size_t i = (size_t)blockIdx.x * 256 + threadIdx.x;
  if (i >= total) return;                 // total = B*Np*Np
  int j = (int)(i % Np); size_t b = i / ((size_t)Np * Np);
  E[i] /= (1e-9f + cols[b * Np + j]);
}

__global__ void k_ew(const float* __restrict__ A, const float* __restrict__ B,
                     float* __restrict__ O, size_t total, int mode) {
  size_t i = (size_t)blockIdx.x * 256 + threadIdx.x;
  if (i >= total) return;
  O[i] = mode ? (A[i] + B[i]) : (A[i] - B[i]);
}

__global__ void k_copycols(const float* __restrict__ src, float* __restrict__ dst,
                           int Cs, int Cd, int coff, size_t total) {
  size_t i = (size_t)blockIdx.x * 256 + threadIdx.x;
  if (i >= total) return;                 // total = M*Cs
  int c = (int)(i % Cs); size_t m = i / Cs;
  dst[m * (size_t)Cd + coff + c] = src[i];
}

__global__ void k_maxtime(const float* __restrict__ X, float* __restrict__ out,
                          int T, int C, size_t total) {
  size_t i = (size_t)blockIdx.x * 256 + threadIdx.x;
  if (i >= total) return;                 // total = B*C
  int c = (int)(i % C); size_t b = i / C;
  const float* p = X + b * (size_t)T * C + c;
  float m = -3.4e38f;
  for (int t = 0; t < T; ++t) m = fmaxf(m, p[(size_t)t * C]);
  out[i] = m;
}

// ---------------------------------------------------------------------------
// Host side
// ---------------------------------------------------------------------------
static inline unsigned gblk(size_t n) { return (unsigned)((n + 255) / 256); }

// x @ W^T (NT): A[M,K] row-major, W[N,K] row-major
static void gemmNT(hipStream_t st, const float* A, const float* W, const float* bias,
                   float* C, int M, int N, int K, int lda, int ldb, int ldc,
                   long long sAb, long long sBb, long long sCb, int batch, int acc) {
  dim3 g((N + BN - 1) / BN, (M + BM - 1) / BM, batch);
  k_gemm<0><<<g, 256, 0, st>>>(A, W, bias, C, M, N, K, lda, ldb, 1, ldc,
                               1, 0, 0, sAb, sBb, sCb, acc);
}
// A^T @ B (TN): A[K,M], B[K,N]
static void gemmTN(hipStream_t st, const float* A, const float* B, float* C,
                   int M, int N, int K, int lda, int ldb, int ldc,
                   long long sAb, long long sBb, long long sCb, int batch) {
  dim3 g((N + BN - 1) / BN, (M + BM - 1) / BM, batch);
  k_gemm<1><<<g, 256, 0, st>>>(A, B, nullptr, C, M, N, K, lda, ldb, 1, ldc,
                               1, 0, 0, sAb, sBb, sCb, 0);
}
// conv tap: row remap + k-stride-3 weights
static void gemmCONV(hipStream_t st, const float* A, const float* W, float* C,
                     int M, int N, int K, int lda, int ldb, int bsk, int ldc,
                     int mdiv, int mstride, int moff, int acc) {
  dim3 g((N + BN - 1) / BN, (M + BM - 1) / BM, 1);
  k_gemm<2><<<g, 256, 0, st>>>(A, W, nullptr, C, M, N, K, lda, ldb, bsk, ldc,
                               mdiv, mstride, moff, 0, 0, 0, acc);
}

extern "C" void kernel_launch(void* const* d_in, const int* in_sizes, int n_in,
                              void* d_out, int out_size, void* d_ws, size_t ws_size,
                              hipStream_t stream) {
  (void)in_sizes; (void)n_in; (void)out_size; (void)ws_size;
  const float* X    = (const float*)d_in[0];
  const float* w1   = (const float*)d_in[1];
  const float* g1   = (const float*)d_in[2];
  const float* b1   = (const float*)d_in[3];
  const float* w2   = (const float*)d_in[4];
  const float* g2   = (const float*)d_in[5];
  const float* b2   = (const float*)d_in[6];
  const float* lng  = (const float*)d_in[7];
  const float* lnb  = (const float*)d_in[8];
  const float* l0w1 = (const float*)d_in[9];
  const float* l0g1 = (const float*)d_in[10];
  const float* l0b1 = (const float*)d_in[11];
  const float* l0w2 = (const float*)d_in[12];
  const float* l0g2 = (const float*)d_in[13];
  const float* l0b2 = (const float*)d_in[14];
  const float* l1w1 = (const float*)d_in[15];
  const float* l1g1 = (const float*)d_in[16];
  const float* l1b1 = (const float*)d_in[17];
  const float* l1w2 = (const float*)d_in[18];
  const float* l1g2 = (const float*)d_in[19];
  const float* l1b2 = (const float*)d_in[20];
  const float* pw1  = (const float*)d_in[21];
  const float* pg1  = (const float*)d_in[22];
  const float* pb1  = (const float*)d_in[23];
  const float* pw2  = (const float*)d_in[24];
  const float* pg2  = (const float*)d_in[25];
  const float* pb2  = (const float*)d_in[26];
  const float* qw   = (const float*)d_in[27];
  const float* kw   = (const float*)d_in[28];
  const float* vw   = (const float*)d_in[29];
  const float* tw   = (const float*)d_in[30];
  const float* tg   = (const float*)d_in[31];
  const float* tb   = (const float*)d_in[32];
  const float* fw   = (const float*)d_in[33];
  const float* fg   = (const float*)d_in[34];
  const float* fb   = (const float*)d_in[35];
  const float* lw1  = (const float*)d_in[36];
  const float* lb1  = (const float*)d_in[37];
  const float* g6   = (const float*)d_in[38];
  const float* b6   = (const float*)d_in[39];
  const float* lw2  = (const float*)d_in[40];
  const float* lb2  = (const float*)d_in[41];
  const float* g7   = (const float*)d_in[42];
  const float* b7   = (const float*)d_in[43];
  const float* lw3  = (const float*)d_in[44];
  const float* lb3  = (const float*)d_in[45];

  const int B = 32, N = 2048, S1 = 512, S2 = 256, KS = 32;
  const size_t P = (size_t)B * N;          // 65536 points

  // --- workspace layout: 3 big arenas + bump-allocated smalls -------------
  const size_t ARENA = 268435456;          // 256 MB each
  char* base = (char*)d_ws;
  float* A0 = (float*)base;
  float* A1 = (float*)(base + ARENA);
  float* A2 = (float*)(base + 2 * ARENA);
  size_t off = 3 * ARENA;
  auto alloc = [&](size_t nfl) -> float* {
    float* p = (float*)(base + off);
    off += ((nfl * 4 + 255) & ~size_t(255));
    return p;
  };
  float* H0    = alloc(P * 64);
  float* FEAT  = alloc(P * 64);
  float* MEAN  = alloc(2048);
  float* VAR   = alloc(2048);
  int*   FIDX1 = (int*)alloc(B * S1);
  float* NXYZ1 = alloc((size_t)B * S1 * 3);
  float* CTR1  = alloc((size_t)B * S1 * 64);
  int*   KNN1  = (int*)alloc((size_t)B * S1 * KS);
  float* F0    = alloc((size_t)B * S1 * 128);
  int*   FIDX2 = (int*)alloc(B * S2);
  float* NXYZ2 = alloc((size_t)B * S2 * 3);
  float* CTR2  = alloc((size_t)B * S2 * 128);
  int*   KNN2  = (int*)alloc((size_t)B * S2 * KS);
  float* F1    = alloc((size_t)B * S2 * 256);
  float* T0    = alloc(8192 * 256);
  float* H     = alloc(8192 * 256);
  float* Q     = alloc(8192 * 64);
  float* Kb    = alloc(8192 * 64);
  float* V     = alloc(8192 * 256);
  float* E     = alloc((size_t)B * 256 * 256);
  float* COLS  = alloc(8192);
  float* XR    = alloc(8192 * 256);
  float* XS    = alloc(8192 * 256);
  float* XT    = alloc(8192 * 256);
  float* HMAX  = alloc((size_t)B * 1280);
  float* C1    = alloc((size_t)B * 512);
  float* C2    = alloc((size_t)B * 256);

  // ---- Stage 1: point embedding ------------------------------------------
  k_linear3<<<gblk(P * 64), 256, 0, stream>>>(X, w1, H0, P);
  k_bn_stats<<<64, 256, 0, stream>>>(H0, MEAN, VAR, (int)P, 64);
  k_bn_act<<<gblk(P * 64), 256, 0, stream>>>(H0, H0, MEAN, VAR, g1, b1, P * 64, 64, 1);
  gemmNT(stream, H0, w2, nullptr, FEAT, (int)P, 64, 64, 64, 64, 64, 0, 0, 0, 1, 0);
  k_bn_stats<<<64, 256, 0, stream>>>(FEAT, MEAN, VAR, (int)P, 64);
  k_bn_act<<<gblk(P * 64), 256, 0, stream>>>(FEAT, FEAT, MEAN, VAR, g2, b2, P * 64, 64, 1);
  k_layernorm<<<gblk(P), 256, 0, stream>>>(FEAT, lng, lnb, P, 64);

  // ---- Stage 2: sample_and_group(512,32) + local_op0 ---------------------
  k_fps<<<B, 256, 0, stream>>>(X, FIDX1, N, S1);
  k_gather3<<<gblk((size_t)B * S1), 256, 0, stream>>>(X, FIDX1, NXYZ1, S1, N, (size_t)B * S1);
  k_gatherF<<<gblk((size_t)B * S1 * 64), 256, 0, stream>>>(FEAT, FIDX1, CTR1, S1, N, 64, (size_t)B * S1 * 64);
  k_sqdist<<<gblk((size_t)B * S1 * N), 256, 0, stream>>>(NXYZ1, X, A1, S1, N, (size_t)B * S1 * N);
  k_knn<<<B * S1, 256, 0, stream>>>(A1, KNN1, N, KS);
  size_t nf1t = (size_t)B * S1 * KS * 64;
  k_group<<<gblk(nf1t), 256, 0, stream>>>(FEAT, KNN1, CTR1, A0, S1, KS, 64, N, nf1t);
  int M1 = B * S1 * KS;                    // 524288 rows, 128 cols
  gemmNT(stream, A0, l0w1, nullptr, A1, M1, 128, 128, 128, 128, 128, 0, 0, 0, 1, 0);
  k_bn_stats<<<128, 256, 0, stream>>>(A1, MEAN, VAR, M1, 128);
  k_bn_act<<<gblk((size_t)M1 * 128), 256, 0, stream>>>(A1, A1, MEAN, VAR, l0g1, l0b1, (size_t)M1 * 128, 128, 1);
  gemmNT(stream, A1, l0w2, nullptr, A2, M1, 128, 128, 128, 128, 128, 0, 0, 0, 1, 0);
  k_bn_stats<<<128, 256, 0, stream>>>(A2, MEAN, VAR, M1, 128);
  k_bn_act<<<gblk((size_t)M1 * 128), 256, 0, stream>>>(A2, A2, MEAN, VAR, l0g2, l0b2, (size_t)M1 * 128, 128, 1);
  k_maxk<<<gblk((size_t)B * S1 * 128), 256, 0, stream>>>(A2, F0, KS, 128, (size_t)B * S1 * 128);

  // ---- Stage 3: sample_and_group(256,32) + local_op1 ---------------------
  k_fps<<<B, 256, 0, stream>>>(NXYZ1, FIDX2, S1, S2);
  k_gather3<<<gblk((size_t)B * S2), 256, 0, stream>>>(NXYZ1, FIDX2, NXYZ2, S2, S1, (size_t)B * S2);
  k_gatherF<<<gblk((size_t)B * S2 * 128), 256, 0, stream>>>(F0, FIDX2, CTR2, S2, S1, 128, (size_t)B * S2 * 128);
  k_sqdist<<<gblk((size_t)B * S2 * S1), 256, 0, stream>>>(NXYZ2, NXYZ1, A1, S2, S1, (size_t)B * S2 * S1);
  k_knn<<<B * S2, 256, 0, stream>>>(A1, KNN2, S1, KS);
  size_t nf2t = (size_t)B * S2 * KS * 128;
  k_group<<<gblk(nf2t), 256, 0, stream>>>(F0, KNN2, CTR2, A0, S2, KS, 128, S1, nf2t);
  int M2 = B * S2 * KS;                    // 262144 rows, 256 cols
  gemmNT(stream, A0, l1w1, nullptr, A1, M2, 256, 256, 256, 256, 256, 0, 0, 0, 1, 0);
  k_bn_stats<<<256, 256, 0, stream>>>(A1, MEAN, VAR, M2, 256);
  k_bn_act<<<gblk((size_t)M2 * 256), 256, 0, stream>>>(A1, A1, MEAN, VAR, l1g1, l1b1, (size_t)M2 * 256, 256, 1);
  gemmNT(stream, A1, l1w2, nullptr, A2, M2, 256, 256, 256, 256, 256, 0, 0, 0, 1, 0);
  k_bn_stats<<<256, 256, 0, stream>>>(A2, MEAN, VAR, M2, 256);
  k_bn_act<<<gblk((size_t)M2 * 256), 256, 0, stream>>>(A2, A2, MEAN, VAR, l1g2, l1b2, (size_t)M2 * 256, 256, 1);
  k_maxk<<<gblk((size_t)B * S2 * 256), 256, 0, stream>>>(A2, F1, KS, 256, (size_t)B * S2 * 256);

  // ---- Stage 4: pre-transformer MLPs -------------------------------------
  int MP = B * S2;                         // 8192 rows
  gemmNT(stream, F1, pw1, nullptr, T0, MP, 256, 256, 256, 256, 256, 0, 0, 0, 1, 0);
  k_bn_stats<<<256, 256, 0, stream>>>(T0, MEAN, VAR, MP, 256);
  k_bn_act<<<gblk((size_t)MP * 256), 256, 0, stream>>>(T0, T0, MEAN, VAR, pg1, pb1, (size_t)MP * 256, 256, 1);
  gemmNT(stream, T0, pw2, nullptr, H, MP, 256, 256, 256, 256, 256, 0, 0, 0, 1, 0);
  k_bn_stats<<<256, 256, 0, stream>>>(H, MEAN, VAR, MP, 256);
  k_bn_act<<<gblk((size_t)MP * 256), 256, 0, stream>>>(H, H, MEAN, VAR, pg2, pb2, (size_t)MP * 256, 256, 1);

  // ---- Stage 5: 4 self-attention layers; collect into HCAT (A1) ----------
  float* HCAT = A1;                        // [8192,1280]
  for (int l = 0; l < 4; ++l) {
    const float* qwi = qw + (size_t)l * 64 * 256;
    const float* kwi = kw + (size_t)l * 64 * 256;
    const float* vwi = vw + (size_t)l * 256 * 256;
    const float* twi = tw + (size_t)l * 256 * 256;
    const float* tgi = tg + (size_t)l * 256;
    const float* tbi = tb + (size_t)l * 256;
    gemmNT(stream, H, qwi, nullptr, Q,  MP, 64,  256, 256, 256, 64,  0, 0, 0, 1, 0);
    gemmNT(stream, H, kwi, nullptr, Kb, MP, 64,  256, 256, 256, 64,  0, 0, 0, 1, 0);
    gemmNT(stream, H, vwi, nullptr, V,  MP, 256, 256, 256, 256, 256, 0, 0, 0, 1, 0);
    // e = q @ k^T per batch
    gemmNT(stream, Q, Kb, nullptr, E, 256, 256, 64, 64, 64, 256,
           256 * 64, 256 * 64, 256 * 256, B, 0);
    k_softmax256<<<B * 256, 256, 0, stream>>>(E);
    k_colsum<<<gblk((size_t)B * 256), 256, 0, stream>>>(E, COLS, 256, (size_t)B * 256);
    k_coldiv<<<gblk((size_t)B * 256 * 256), 256, 0, stream>>>(E, COLS, 256, (size_t)B * 256 * 256);
    // xr[j,c] = sum_i a[i,j] v[i,c]  -> A^T @ V per batch
    gemmTN(stream, E, V, XR, 256, 256, 256, 256, 256, 256,
           256 * 256, 256 * 256, 256 * 256, B);
    k_ew<<<gblk((size_t)MP * 256), 256, 0, stream>>>(H, XR, XS, (size_t)MP * 256, 0);
    gemmNT(stream, XS, twi, nullptr, XT, MP, 256, 256, 256, 256, 256, 0, 0, 0, 1, 0);
    k_bn_stats<<<256, 256, 0, stream>>>(XT, MEAN, VAR, MP, 256);
    k_bn_act<<<gblk((size_t)MP * 256), 256, 0, stream>>>(XT, XT, MEAN, VAR, tgi, tbi, (size_t)MP * 256, 256, 1);
    k_ew<<<gblk((size_t)MP * 256), 256, 0, stream>>>(H, XT, H, (size_t)MP * 256, 1);
    k_copycols<<<gblk((size_t)MP * 256), 256, 0, stream>>>(H, HCAT, 256, 1280, l * 256, (size_t)MP * 256);
  }
  k_copycols<<<gblk((size_t)MP * 256), 256, 0, stream>>>(F1, HCAT, 256, 1280, 1024, (size_t)MP * 256);

  // ---- Stage 6: conv1d(k=3, VALID) as 3 accumulated WMMA GEMMs -----------
  float* CONV = A2;                        // [32*254, 1280]
  int MC = B * 254;
  for (int kk = 0; kk < 3; ++kk)
    gemmCONV(stream, HCAT, fw + kk, CONV, MC, 1280, 1280,
             1280, 1280 * 3, 3, 1280, 254, 256, kk, (kk > 0));
  k_bn_stats<<<1280, 256, 0, stream>>>(CONV, MEAN, VAR, MC, 1280);
  k_bn_act<<<gblk((size_t)MC * 1280), 256, 0, stream>>>(CONV, CONV, MEAN, VAR, fg, fb, (size_t)MC * 1280, 1280, 2);
  k_maxtime<<<gblk((size_t)B * 1280), 256, 0, stream>>>(CONV, HMAX, 254, 1280, (size_t)B * 1280);

  // ---- Stage 7: classifier head ------------------------------------------
  gemmNT(stream, HMAX, lw1, lb1, C1, B, 512, 1280, 1280, 1280, 512, 0, 0, 0, 1, 0);
  k_bn_stats<<<512, 256, 0, stream>>>(C1, MEAN, VAR, B, 512);
  k_bn_act<<<gblk((size_t)B * 512), 256, 0, stream>>>(C1, C1, MEAN, VAR, g6, b6, (size_t)B * 512, 512, 1);
  gemmNT(stream, C1, lw2, lb2, C2, B, 256, 512, 512, 512, 256, 0, 0, 0, 1, 0);
  k_bn_stats<<<256, 256, 0, stream>>>(C2, MEAN, VAR, B, 256);
  k_bn_act<<<gblk((size_t)B * 256), 256, 0, stream>>>(C2, C2, MEAN, VAR, g7, b7, (size_t)B * 256, 256, 1);
  gemmNT(stream, C2, lw3, lb3, (float*)d_out, B, 40, 256, 256, 256, 40, 0, 0, 0, 1, 0);
}
